// Attention_78048145703090
// MI455X (gfx1250) — matrix operand order repaired and checked by reading, not compiled, not run on hardware
//
#include <hip/hip_runtime.h>
#include <hip/hip_bf16.h>

typedef __attribute__((ext_vector_type(16))) _Float16 v16h;
typedef __attribute__((ext_vector_type(8)))  _Float16 v8h;
typedef __attribute__((ext_vector_type(8)))  float    v8f;

#define BATCH 8
#define NPOS  4096    // H*W = 64*64
#define CH    64
#define CENTER 2080   // (H/2)*W + W/2 = 32*64+32

// ---------------------------------------------------------------------------
// A-fragment loader for V_WMMA_F32_16X16X32_F16.
// 16-bit A 16x32, this code's k order (not the ISA table's): lane's row M = lane&15, kh = lane>>4.
//   halfs i=0..15 -> K = 16*kh + i       (contiguous, matches B loads)
// ---------------------------------------------------------------------------
__device__ inline v16h load_a_frag(const _Float16* base, int kh) {
    v8h lo = *(const v8h*)(base + 16 * kh);
    v8h hi = *(const v8h*)(base + 16 * kh + 8);
    v16h r;
#pragma unroll
    for (int i = 0; i < 8; ++i) { r[i] = lo[i]; r[i + 8] = hi[i]; }
    return r;
}

// ---------------------------------------------------------------------------
// f32 -> f16 conversion (x, wk, wv, wp)
// ---------------------------------------------------------------------------
__global__ void cvt_f32_f16(const float* __restrict__ src, _Float16* __restrict__ dst, int n) {
    int i = blockIdx.x * blockDim.x + threadIdx.x;
    if (i < n) dst[i] = (_Float16)src[i];
}

// ---------------------------------------------------------------------------
// central[b,c] = bq[c] + sum_j x[b,CENTER,j] * wq[c,j]   (only q-row we need)
// ---------------------------------------------------------------------------
__global__ void central_kernel(const float* __restrict__ x, const float* __restrict__ wq,
                               const float* __restrict__ bq, float* __restrict__ central) {
    int b = blockIdx.x, c = threadIdx.x;  // 8 x 64
    const float* xr = x + ((size_t)b * NPOS + CENTER) * CH;
    const float* wr = wq + c * CH;
    float acc = bq[c];
#pragma unroll 8
    for (int j = 0; j < CH; ++j) acc += xr[j] * wr[j];
    central[b * CH + c] = acc;
}

// ---------------------------------------------------------------------------
// K = x@wk^T + bk (f32, for score dot)  and  V^T = (x@wv^T + bv)^T (f16 [B,C,N])
// One wave computes a 16-row x 64-col strip of both via WMMA.
// ---------------------------------------------------------------------------
__global__ __launch_bounds__(32) void proj_kv_kernel(
    const _Float16* __restrict__ Xh, const _Float16* __restrict__ Wkh,
    const _Float16* __restrict__ Wvh, const float* __restrict__ bk,
    const float* __restrict__ bv, float* __restrict__ Kf, _Float16* __restrict__ Vt) {
    int lane = threadIdx.x;
    int col  = lane & 15;   // N index of B/D frags; also M row of A frag
    int kh   = lane >> 4;
    int bid  = blockIdx.x;              // 0..2047
    int b    = bid >> 8;
    int n0   = (bid & 255) << 4;

    const _Float16* arow = Xh + ((size_t)b * NPOS + n0 + col) * CH;

    v8f z = {};
    v8f accK[4] = {z, z, z, z};
    v8f accV[4] = {z, z, z, z};

#pragma unroll
    for (int ks = 0; ks < 2; ++ks) {            // K = 64 = 2 x 32
        int kbase = ks * 32;
        v16h a = load_a_frag(arow + kbase, kh);
#pragma unroll
        for (int ct = 0; ct < 4; ++ct) {
            int c = ct * 16 + col;
            // B[k,n] = w[c0+n, kbase+k]; per-lane 16 contiguous halfs (32B aligned)
            v16h bkf = *(const v16h*)(Wkh + c * CH + kbase + 16 * kh);
            accK[ct] = __builtin_amdgcn_wmma_f32_16x16x32_f16(
                false, a, false, bkf, (short)0, accK[ct], false, false);
            v16h bvf = *(const v16h*)(Wvh + c * CH + kbase + 16 * kh);
            accV[ct] = __builtin_amdgcn_wmma_f32_16x16x32_f16(
                false, a, false, bvf, (short)0, accV[ct], false, false);
        }
    }

#pragma unroll
    for (int ct = 0; ct < 4; ++ct) {
        int c = ct * 16 + col;
        float bkc = bk[c], bvc = bv[c];
#pragma unroll
        for (int r = 0; r < 8; ++r) {           // D: row M = r + 8*kh, col N = col
            int m = r + 8 * kh;
            Kf[((size_t)b * NPOS + n0 + m) * CH + c] = accK[ct][r] + bkc;
        }
        v8h vst;                                 // V^T: 8 contiguous m's per lane
#pragma unroll
        for (int r = 0; r < 8; ++r) vst[r] = (_Float16)(accV[ct][r] + bvc);
        *(v8h*)(Vt + ((size_t)b * CH + c) * NPOS + n0 + 8 * kh) = vst;
    }
}

// ---------------------------------------------------------------------------
// s[b,m] = scale * central[b] . K[b,m]
// ---------------------------------------------------------------------------
__global__ void score_kernel(const float* __restrict__ central, const float* __restrict__ Kf,
                             float* __restrict__ s) {
    int t = blockIdx.x * blockDim.x + threadIdx.x;  // 0..32767
    int b = t >> 12;
    const float* c0 = central + b * CH;
    const float* kr = Kf + (size_t)t * CH;
    float acc = 0.f;
#pragma unroll 8
    for (int j = 0; j < CH; ++j) acc += c0[j] * kr[j];
    s[t] = acc * 0.125f;   // scale = 64^-0.5
}

// ---------------------------------------------------------------------------
// smax[b] = max_m s[b,m]
// ---------------------------------------------------------------------------
__global__ void smax_kernel(const float* __restrict__ s, float* __restrict__ smax) {
    __shared__ float red[256];
    int b = blockIdx.x, tid = threadIdx.x;
    float m = -3.0e38f;
    for (int i = tid; i < NPOS; i += 256) m = fmaxf(m, s[b * NPOS + i]);
    red[tid] = m;
    __syncthreads();
    for (int st = 128; st > 0; st >>= 1) {
        if (tid < st) red[tid] = fmaxf(red[tid], red[tid + st]);
        __syncthreads();
    }
    if (tid == 0) smax[b] = red[0];
}

// ---------------------------------------------------------------------------
// Fused softmax(P) @ V without materializing P:
//   P[n,m] = exp(weights[n] * (s[b,m] - smax[b]))   (weights[n] > 0)
//   out[n,c] = (sum_m P[n,m] * V[m,c]) / (sum_m P[n,m])
// One wave owns 16 output rows x all 64 cols. Software-pipelined: step i+1's
// s-vector and V-fragment loads are issued before step i's exp+WMMA compute,
// so global latency overlaps the 16 v_exp + 4 v_wmma of each step.
// ---------------------------------------------------------------------------
__global__ __launch_bounds__(32) void attn_kernel(
    const float* __restrict__ s, const float* __restrict__ smax,
    const _Float16* __restrict__ Vt, _Float16* __restrict__ Oh) {
    __shared__ float rsum[16];
    int lane = threadIdx.x;
    int col  = lane & 15;
    int kh   = lane >> 4;
    int bid  = blockIdx.x;
    int b    = bid >> 8;
    int n0   = (bid & 255) << 4;

    // row weight for this lane's A-fragment row (M = lane&15)
    int n = n0 + col;
    float dy = (float)(n >> 6) - 32.0f;
    float dx = (float)(n & 63) - 32.0f;
    float wrow  = __expf(1.0f - sqrtf(dy * dy + dx * dx));
    float smaxb = smax[b];

    const float* sb = s + (size_t)b * NPOS;
    const _Float16* vb = Vt + (size_t)b * CH * NPOS;

    v8f z = {};
    v8f acc[4] = {z, z, z, z};
    float rs = 0.f;

    const int STEPS = NPOS / 32;   // 128

    // --- prologue: load step 0 ---
    const float4* plo = (const float4*)(sb + 16 * kh);
    const float4* phi = (const float4*)(sb + 16 * kh + 8);
    float4 c0 = plo[0], c1 = plo[1], c2 = phi[0], c3 = phi[1];
    v16h cb0 = *(const v16h*)(vb + (0 * 16 + col) * NPOS + 16 * kh);
    v16h cb1 = *(const v16h*)(vb + (1 * 16 + col) * NPOS + 16 * kh);
    v16h cb2 = *(const v16h*)(vb + (2 * 16 + col) * NPOS + 16 * kh);
    v16h cb3 = *(const v16h*)(vb + (3 * 16 + col) * NPOS + 16 * kh);

    for (int it = 0; it < STEPS; ++it) {
        // --- issue next step's loads (clamped on last iteration; uniform) ---
        int mn = (it + 1 < STEPS) ? (it + 1) * 32 : it * 32;
        const float4* nlo = (const float4*)(sb + mn + 16 * kh);
        const float4* nhi = (const float4*)(sb + mn + 16 * kh + 8);
        float4 x0 = nlo[0], x1 = nlo[1], x2 = nhi[0], x3 = nhi[1];
        v16h nb0 = *(const v16h*)(vb + (0 * 16 + col) * NPOS + mn + 16 * kh);
        v16h nb1 = *(const v16h*)(vb + (1 * 16 + col) * NPOS + mn + 16 * kh);
        v16h nb2 = *(const v16h*)(vb + (2 * 16 + col) * NPOS + mn + 16 * kh);
        v16h nb3 = *(const v16h*)(vb + (3 * 16 + col) * NPOS + mn + 16 * kh);

        // --- compute current step (all operands already in registers) ---
        float sv[16] = {c0.x, c0.y, c0.z, c0.w, c1.x, c1.y, c1.z, c1.w,
                        c2.x, c2.y, c2.z, c2.w, c3.x, c3.y, c3.z, c3.w};
        v16h a;
#pragma unroll
        for (int i = 0; i < 16; ++i) {
            float e = __expf(wrow * (sv[i] - smaxb));  // <= 1, safe
            rs += e;
            a[i] = (_Float16)e;
        }
        acc[0] = __builtin_amdgcn_wmma_f32_16x16x32_f16(false, a, false, cb0, (short)0, acc[0], false, false);
        acc[1] = __builtin_amdgcn_wmma_f32_16x16x32_f16(false, a, false, cb1, (short)0, acc[1], false, false);
        acc[2] = __builtin_amdgcn_wmma_f32_16x16x32_f16(false, a, false, cb2, (short)0, acc[2], false, false);
        acc[3] = __builtin_amdgcn_wmma_f32_16x16x32_f16(false, a, false, cb3, (short)0, acc[3], false, false);

        // --- rotate buffers ---
        c0 = x0; c1 = x1; c2 = x2; c3 = x3;
        cb0 = nb0; cb1 = nb1; cb2 = nb2; cb3 = nb3;
    }

    // lanes L and L+16 cover complementary m's of row L
    float rsAll = rs + __shfl_xor(rs, 16);
    if (lane < 16) rsum[lane] = rsAll;
    __syncthreads();

    float inv[8];
#pragma unroll
    for (int r = 0; r < 8; ++r) inv[r] = 1.0f / rsum[r + 8 * kh];

#pragma unroll
    for (int ct = 0; ct < 4; ++ct) {
#pragma unroll
        for (int r = 0; r < 8; ++r) {
            Oh[((size_t)b * NPOS + n0 + r + 8 * kh) * CH + ct * 16 + col] =
                (_Float16)(acc[ct][r] * inv[r]);
        }
    }
}

// ---------------------------------------------------------------------------
// out = attn_out @ wp^T + bp   (f32 result to d_out)
// ---------------------------------------------------------------------------
__global__ __launch_bounds__(32) void out_proj_kernel(
    const _Float16* __restrict__ Oh, const _Float16* __restrict__ Wph,
    const float* __restrict__ bp, float* __restrict__ out) {
    int lane = threadIdx.x;
    int col  = lane & 15;
    int kh   = lane >> 4;
    int bid  = blockIdx.x;
    int b    = bid >> 8;
    int n0   = (bid & 255) << 4;

    const _Float16* arow = Oh + ((size_t)b * NPOS + n0 + col) * CH;

    v8f z = {};
    v8f acc[4] = {z, z, z, z};
#pragma unroll
    for (int ks = 0; ks < 2; ++ks) {
        int kbase = ks * 32;
        v16h a = load_a_frag(arow + kbase, kh);
#pragma unroll
        for (int ct = 0; ct < 4; ++ct) {
            int c = ct * 16 + col;
            v16h bf = *(const v16h*)(Wph + c * CH + kbase + 16 * kh);
            acc[ct] = __builtin_amdgcn_wmma_f32_16x16x32_f16(
                false, a, false, bf, (short)0, acc[ct], false, false);
        }
    }
#pragma unroll
    for (int ct = 0; ct < 4; ++ct) {
        int c = ct * 16 + col;
        float bpc = bp[c];
#pragma unroll
        for (int r = 0; r < 8; ++r) {
            out[((size_t)b * NPOS + n0 + r + 8 * kh) * CH + c] = acc[ct][r] + bpc;
        }
    }
}

// ---------------------------------------------------------------------------
extern "C" void kernel_launch(void* const* d_in, const int* in_sizes, int n_in,
                              void* d_out, int out_size, void* d_ws, size_t ws_size,
                              hipStream_t stream) {
    (void)in_sizes; (void)n_in; (void)out_size; (void)ws_size;
    const float* x  = (const float*)d_in[0];
    const float* wq = (const float*)d_in[1];
    const float* bq = (const float*)d_in[2];
    const float* wk = (const float*)d_in[3];
    const float* bk = (const float*)d_in[4];
    const float* wv = (const float*)d_in[5];
    const float* bv = (const float*)d_in[6];
    const float* wp = (const float*)d_in[7];
    const float* bp = (const float*)d_in[8];

    char* w = (char*)d_ws;
    const size_t NELEM = (size_t)BATCH * NPOS * CH;            // 2,097,152
    float*    Kf      = (float*)   (w + 0);                    // 8 MB
    _Float16* Vt      = (_Float16*)(w + 8388608);              // 4 MB  [B,C,N] f16
    _Float16* Xh      = (_Float16*)(w + 12582912);             // 4 MB
    _Float16* Oh      = (_Float16*)(w + 16777216);             // 4 MB
    _Float16* Wkh     = (_Float16*)(w + 20971520);             // 8 KB
    _Float16* Wvh     = (_Float16*)(w + 20979712);             // 8 KB
    _Float16* Wph     = (_Float16*)(w + 20987904);             // 8 KB
    float*    central = (float*)   (w + 20996096);             // 2 KB
    float*    s       = (float*)   (w + 20998144);             // 128 KB
    float*    smax    = (float*)   (w + 21129216);             // 32 B

    cvt_f32_f16<<<(int)(NELEM / 256), 256, 0, stream>>>(x, Xh, (int)NELEM);
    cvt_f32_f16<<<16, 256, 0, stream>>>(wk, Wkh, CH * CH);
    cvt_f32_f16<<<16, 256, 0, stream>>>(wv, Wvh, CH * CH);
    cvt_f32_f16<<<16, 256, 0, stream>>>(wp, Wph, CH * CH);

    central_kernel<<<BATCH, CH, 0, stream>>>(x, wq, bq, central);

    proj_kv_kernel<<<BATCH * (NPOS / 16), 32, 0, stream>>>(Xh, Wkh, Wvh, bk, bv, Kf, Vt);

    score_kernel<<<(BATCH * NPOS) / 256, 256, 0, stream>>>(central, Kf, s);
    smax_kernel<<<BATCH, 256, 0, stream>>>(s, smax);

    attn_kernel<<<BATCH * (NPOS / 16), 32, 0, stream>>>(s, smax, Vt, Oh);

    out_proj_kernel<<<BATCH * (NPOS / 16), 32, 0, stream>>>(Oh, Wph, bp, (float*)d_out);
}
